// CRF_17068200034336
// MI455X (gfx1250) — matrix operand order repaired and checked by reading, not compiled, not run on hardware
//
#include <hip/hip_runtime.h>
#include <hip/hip_bf16.h>
#include <stdint.h>

// CRF NLL, forward scan as exp-domain GEMM on the WMMA units.
// B=256, S=256, T=20 (padded to 32). One wave per block, 16 batches per wave.
// Per step: 2x v_wmma_f32_16x16x32_f16, raw v_exp/v_log transcendentals,
// v_permlanex16 half-swaps (no LDS), branchless masked update.

typedef __attribute__((ext_vector_type(16))) _Float16 v16h;
typedef __attribute__((ext_vector_type(8)))  float    v8f;

#define Bsz 256
#define Ssz 256
#define Tt  20
#define TP  32
#define START_TAG 18
#define STOP_TAG  19

#define LOG2E 1.4426950408889634f
#define LN2   0.6931471805599453f

union V16U { uint32_t u[8]; v16h v; };
union D8U  { v8f v; float f[8]; };

// xor-16 half swap across wave32: single v_permlanex16_b32 (identity selects).
__device__ __forceinline__ uint32_t xhalf_u32(uint32_t x) {
    return __builtin_amdgcn_permlanex16(x, x, 0x76543210u, 0xfedcba98u, false, false);
}
__device__ __forceinline__ float xhalf_f32(float x) {
    return __uint_as_float(xhalf_u32(__float_as_uint(x)));
}

// pack two f32 -> one u32 of f16 pair (v_cvt_pkrtz_f16_f32)
__device__ __forceinline__ uint32_t pk_h2(float a, float b) {
    return __builtin_bit_cast(uint32_t, __builtin_amdgcn_cvt_pkrtz(a, b));
}

// ---------------------------------------------------------------------------
// Prep: Ehat[j][i] = exp(trans[i][j]) as f16, zero-padded to 32x32.
// exp(-10000) underflows to 0, killing the START column / STOP row.
// ---------------------------------------------------------------------------
__global__ void crf_prep(const float* __restrict__ trans,
                         _Float16* __restrict__ Ehat) {
    int idx = blockIdx.x * blockDim.x + threadIdx.x;   // 0..1023
    if (idx >= TP * TP) return;
    int j = idx >> 5, i = idx & 31;
    float v = 0.f;
    if (i < Tt && j < Tt) v = __expf(trans[i * Tt + j]);
    Ehat[idx] = (_Float16)v;
}

// ---------------------------------------------------------------------------
// Forward scan. Partition P[j][b]: j across VGPR index (+half), b = lane & 15.
//   tile0 (p0[k]): j = k + 8*half
//   tile1 (p1[k]): j = 16 + k (half 0, k<4 valid), else -1e30 padding
// Matches WMMA C/D layout with M=j (state), N=b (batch).
// ---------------------------------------------------------------------------
__global__ __launch_bounds__(32) void crf_scan(
    const float* __restrict__ feats, const float* __restrict__ trans,
    const int* __restrict__ mask, const _Float16* __restrict__ Ehat,
    float* __restrict__ fwdPart) {

    const int lane = threadIdx.x & 31;
    const int half = lane >> 4;
    const int lb   = lane & 15;
    const int b    = blockIdx.x * 16 + lb;

    // ---- A operands (loop-invariant): A1 = Ehat rows 0..15, A2 = rows 16..31.
    // A as loaded here (contiguous K, same order as B): lane%16 = M; lanes<16 hold K 0..15;
    // lanes>=16 hold K 16..31. Rows are 64B; 16B-aligned uint4 loads.
    const uint4* E16 = (const uint4*)Ehat;
    const int g  = 2 * half;      // K 0..7  vs 16..23
    const int gh = 2 * half + 1;  // K 8..15 vs 24..31
    uint4 a1l = E16[lb * 4 + g],        a1h = E16[lb * 4 + gh];
    uint4 a2l = E16[(16 + lb) * 4 + g], a2h = E16[(16 + lb) * 4 + gh];
    V16U A1, A2;
    A1.u[0]=a1l.x; A1.u[1]=a1l.y; A1.u[2]=a1l.z; A1.u[3]=a1l.w;
    A1.u[4]=a1h.x; A1.u[5]=a1h.y; A1.u[6]=a1h.z; A1.u[7]=a1h.w;
    A2.u[0]=a2l.x; A2.u[1]=a2l.y; A2.u[2]=a2l.z; A2.u[3]=a2l.w;
    A2.u[4]=a2h.x; A2.u[5]=a2h.y; A2.u[6]=a2h.z; A2.u[7]=a2h.w;

    const float4* F4 = (const float4*)feats;

    // ---- sequence length (mask rows are prefix masks: mask[b,s] = s < len[b])
    int len = 0;
    {
        const int4* M4 = (const int4*)mask;
        const int base = (b * Ssz) >> 2;
#pragma unroll 4
        for (int t = 0; t < Ssz / 4; t++) {
            int4 mv = M4[base + t];
            len += mv.x + mv.y + mv.z + mv.w;
        }
    }

    // ---- init: partition = feats[b,0,:] + trans[START,:]
    float p0[8], p1[8];
    {
        size_t eb = ((size_t)b * Ssz) * Tt;
        float4 L0 = F4[(eb >> 2) + (half ? 2 : 0)];  // j 0..3  / 8..11
        float4 L1 = F4[(eb >> 2) + (half ? 3 : 1)];  // j 4..7  / 12..15
        float4 L2 = F4[(eb >> 2) + 4];               // j 16..19 (in-bounds)
        float f0[8] = {L0.x,L0.y,L0.z,L0.w,L1.x,L1.y,L1.z,L1.w};
        float f2[4] = {L2.x,L2.y,L2.z,L2.w};
#pragma unroll
        for (int k = 0; k < 8; k++)
            p0[k] = f0[k] + trans[START_TAG * Tt + k + half * 8];
#pragma unroll
        for (int k = 0; k < 8; k++)
            p1[k] = (half == 0 && k < 4)
                  ? f2[k] + trans[START_TAG * Tt + 16 + k] : -1e30f;
    }

    // ---- software-pipelined feats row (row s used in step s)
    float4 C0, C1, C2;
    {
        size_t eb = ((size_t)b * Ssz + 1) * Tt;
        C0 = F4[(eb >> 2) + (half ? 2 : 0)];
        C1 = F4[(eb >> 2) + (half ? 3 : 1)];
        C2 = F4[(eb >> 2) + 4];
    }

    const v8f zacc = {0.f,0.f,0.f,0.f,0.f,0.f,0.f,0.f};

#pragma unroll 1
    for (int s = 1; s < Ssz; s++) {
        // issue next row's loads early (clamped), plus a deep prefetch
        int sn = (s + 1 < Ssz) ? s + 1 : s;
        size_t ebn = ((size_t)b * Ssz + sn) * Tt;
        float4 N0 = F4[(ebn >> 2) + (half ? 2 : 0)];
        float4 N1 = F4[(ebn >> 2) + (half ? 3 : 1)];
        float4 N2 = F4[(ebn >> 2) + 4];
        int sp = (s + 8 < Ssz) ? s + 8 : Ssz - 1;
        __builtin_prefetch(feats + ((size_t)b * Ssz + sp) * Tt, 0, 3);

        // 1) per-batch max m[b]: per-lane over VGPRs, combine halves (permlanex16)
        float m = p0[0];
#pragma unroll
        for (int k = 1; k < 8; k++) m = fmaxf(m, p0[k]);
#pragma unroll
        for (int k = 0; k < 8; k++) m = fmaxf(m, p1[k]);
        m = fmaxf(m, xhalf_f32(m));

        // 2) pexp = exp2((P-m)*log2e) in f32, pack to f16 pairs (cvt_pkrtz)
        float e0[8], e1[8];
#pragma unroll
        for (int k = 0; k < 8; k++)
            e0[k] = __builtin_amdgcn_exp2f((p0[k] - m) * LOG2E);
#pragma unroll
        for (int k = 0; k < 8; k++)
            e1[k] = __builtin_amdgcn_exp2f((p1[k] - m) * LOG2E);
        uint32_t uL[4], uH[4];
#pragma unroll
        for (int w = 0; w < 4; w++) {
            uL[w] = pk_h2(e0[2*w], e0[2*w+1]);
            uH[w] = pk_h2(e1[2*w], e1[2*w+1]);
        }

        // 3) B operand (32x16): lanes<16 K0..15 = [own eL | partner eL],
        //    lanes>=16 K16..31 = [partner eH | own eH]. One cross value per word.
        V16U Bm;
#pragma unroll
        for (int w = 0; w < 4; w++) {
            uint32_t t  = half ? uL[w] : uH[w];   // value sent across halves
            uint32_t ct = xhalf_u32(t);
            Bm.u[w]     = half ? ct    : uL[w];
            Bm.u[4 + w] = half ? uH[w] : ct;
        }

        // 4) raw[j][b] = sum_i exp(trans[i][j]) * pexp[i][b]  (two WMMAs)
        v8f d0 = __builtin_amdgcn_wmma_f32_16x16x32_f16(
            false, A1.v, false, Bm.v, (short)0, zacc, false, false);
        v8f d1 = __builtin_amdgcn_wmma_f32_16x16x32_f16(
            false, A2.v, false, Bm.v, (short)0, zacc, false, false);
        D8U D0, D1; D0.v = d0; D1.v = d1;

        // 5) P' = feats + m + ln2*log2(raw); branchless select on s < len[b]
        bool mk = s < len;
        float f0[8] = {C0.x,C0.y,C0.z,C0.w,C1.x,C1.y,C1.z,C1.w};
        float f2[4] = {C2.x,C2.y,C2.z,C2.w};
#pragma unroll
        for (int k = 0; k < 8; k++) {
            float pn = f0[k] + m + LN2 * __builtin_amdgcn_logf(D0.f[k]);
            p0[k] = mk ? pn : p0[k];
        }
#pragma unroll
        for (int k = 0; k < 4; k++) {
            float pn = f2[k] + m + LN2 * __builtin_amdgcn_logf(D1.f[k]);
            p1[k] = (mk && half == 0) ? pn : p1[k];
        }

        C0 = N0; C1 = N1; C2 = N2;
    }

    // ---- final[b] = logsumexp_j( P[j][b] + trans[j][STOP] )
    float q[16];
#pragma unroll
    for (int k = 0; k < 8; k++)
        q[k] = p0[k] + trans[(k + half * 8) * Tt + STOP_TAG];
#pragma unroll
    for (int k = 0; k < 8; k++)
        q[8 + k] = (half == 0 && k < 4)
                 ? p1[k] + trans[(16 + k) * Tt + STOP_TAG] : -1e30f;
    float m2 = q[0];
#pragma unroll
    for (int k = 1; k < 16; k++) m2 = fmaxf(m2, q[k]);
    m2 = fmaxf(m2, xhalf_f32(m2));
    float ss = 0.f;
#pragma unroll
    for (int k = 0; k < 16; k++)
        ss += __builtin_amdgcn_exp2f((q[k] - m2) * LOG2E);
    ss += xhalf_f32(ss);
    float fin = LN2 * __builtin_amdgcn_logf(ss) + m2;

    // sum over the 16 batches of this tile (halves duplicate -> zero one)
    float red = half ? 0.f : fin;
#pragma unroll
    for (int msk = 8; msk >= 1; msk >>= 1) red += __shfl_xor(red, msk, 32);
    if (lane == 0) fwdPart[blockIdx.x] = red;
}

// ---------------------------------------------------------------------------
// Gold path: tg_sum + end_energy, one thread per batch (trivial work).
// ---------------------------------------------------------------------------
__global__ void crf_gold(const float* __restrict__ feats,
                         const float* __restrict__ trans,
                         const int* __restrict__ mask,
                         const int* __restrict__ tags,
                         float* __restrict__ gold) {
    int b = blockIdx.x * blockDim.x + threadIdx.x;
    if (b >= Bsz) return;
    float acc = 0.f;
    int prev = START_TAG, lastTag = 0;
    for (int s = 0; s < Ssz; s++) {
        int mk  = mask[b * Ssz + s];
        int cur = tags[b * Ssz + s];
        float sc = feats[((size_t)b * Ssz + s) * Tt + cur] + trans[prev * Tt + cur];
        acc     = mk ? acc + sc : acc;
        lastTag = mk ? cur : lastTag;
        prev = cur;
    }
    acc += trans[lastTag * Tt + STOP_TAG];   // end_energy (len >= 1 always)
    gold[b] = acc;
}

// ---------------------------------------------------------------------------
// Deterministic finalize: out = sum(fwd partials) - sum(gold).
// ---------------------------------------------------------------------------
__global__ void crf_final(const float* __restrict__ fwdPart,
                          const float* __restrict__ gold,
                          float* __restrict__ out) {
    if (threadIdx.x == 0 && blockIdx.x == 0) {
        float f = 0.f;
        for (int i = 0; i < 16; i++) f += fwdPart[i];
        float g = 0.f;
        for (int i = 0; i < Bsz; i++) g += gold[i];
        out[0] = f - g;
    }
}

extern "C" void kernel_launch(void* const* d_in, const int* in_sizes, int n_in,
                              void* d_out, int out_size, void* d_ws, size_t ws_size,
                              hipStream_t stream) {
    const float* feats = (const float*)d_in[0];   // (256,256,20) f32
    const float* trans = (const float*)d_in[1];   // (20,20) f32
    const int*   mask  = (const int*)d_in[2];     // (256,256) i32
    const int*   tags  = (const int*)d_in[3];     // (256,256) i32
    float* out = (float*)d_out;

    char* ws = (char*)d_ws;
    _Float16* Ehat = (_Float16*)ws;            // 32*32*2  = 2048 B
    float* fwdPart = (float*)(ws + 2048);      // 16*4     =   64 B
    float* gold    = (float*)(ws + 2048 + 64); // 256*4    = 1024 B

    crf_prep <<<dim3(1),  dim3(1024), 0, stream>>>(trans, Ehat);
    crf_scan <<<dim3(16), dim3(32),   0, stream>>>(feats, trans, mask, Ehat, fwdPart);
    crf_gold <<<dim3(2),  dim3(128),  0, stream>>>(feats, trans, mask, tags, gold);
    crf_final<<<dim3(1),  dim3(1),    0, stream>>>(fwdPart, gold, out);
}
